// stagn_2d_model_8022998909567
// MI455X (gfx1250) — matrix operand order repaired and checked by reading, not compiled, not run on hardware
//
#include <hip/hip_runtime.h>
#include <hip/hip_bf16.h>

// ---------------------------------------------------------------------------
// STAGN model for MI455X (gfx1250, wave32, WMMA).
// Dominant cost: conv_out(4096x65536) @ fc1_W(65536x256) = 137 GFLOP.
//  - fc1_W converted once to f16 [N][K] (32 MB -> resident in 192MB L2).
//  - conv(2x2,SAME)+ReLU fused into A-operand production in LDS (no 1 GB
//    conv_out round-trip through HBM).
//  - v_wmma_f32_16x16x32_f16, f32 accumulation, fused bias+ReLU epilogue.
// Everything else (attention, graph convs, edge MLP, head) is negligible
// FLOPs and implemented as simple kernels.
// ---------------------------------------------------------------------------

typedef __attribute__((ext_vector_type(16))) _Float16 v16h;
typedef __attribute__((ext_vector_type(8)))  _Float16 v8h;
typedef __attribute__((ext_vector_type(8)))  float    v8f;

#define BB      4096      // batch
#define TT      8
#define FF      64
#define NN      2048      // nodes
#define DE      128
#define NFC     64        // conv channels
#define FLAT    65536     // NFC*TT*2*FF
#define HID1    256
#define HID2    24

// ---------------------------------------------------------------------------
// 0) zero scratch (degree + aggregation buffers used with atomics)
// ---------------------------------------------------------------------------
__global__ void k_zero(float* __restrict__ p, int n) {
  int i = blockIdx.x * blockDim.x + threadIdx.x;
  if (i < n) p[i] = 0.f;
}

// ---------------------------------------------------------------------------
// 1) attention: per-batch 8x8 additive attention -> att_out (f16, [B][8][128])
//    att[:, :, 0:64] = X ; att[:, :, 64:128] = ctx
//    4 waves / block, one batch item per wave. ~0.6 GFLOP total.
// ---------------------------------------------------------------------------
#define AW 4
__global__ void k_attention(const float* __restrict__ X,
                            const float* __restrict__ Wm,
                            const float* __restrict__ Um,
                            const float* __restrict__ Vm,
                            _Float16* __restrict__ att_h) {
  __shared__ float sW[64 * 64];
  __shared__ float sU[64 * 64];
  __shared__ float sV[64];
  __shared__ float sX[AW][512];
  __shared__ float sa[AW][512];
  __shared__ float su[AW][512];
  __shared__ float sE[AW][64];
  __shared__ float sAl[AW][64];

  const int tid = threadIdx.x;
  for (int i = tid; i < 64 * 64; i += 128) { sW[i] = Wm[i]; sU[i] = Um[i]; }
  if (tid < 64) sV[tid] = Vm[tid];

  const int w = tid >> 5, lane = tid & 31;
  const int b = blockIdx.x * AW + w;

  for (int i = lane; i < 512; i += 32) sX[w][i] = X[(size_t)b * 512 + i];
  __syncthreads();

  // a = X @ W, u = X @ U   (8x64 each)
  for (int i = lane; i < 512; i += 32) {
    const int t = i >> 6, h = i & 63;
    const float* xr = &sX[w][t * 64];
    float aA = 0.f, aU = 0.f;
    for (int f = 0; f < 64; ++f) { aA += xr[f] * sW[f * 64 + h]; aU += xr[f] * sU[f * 64 + h]; }
    sa[w][i] = aA; su[w][i] = aU;
  }
  __syncthreads();

  // e[i][j] = sum_h tanh(a[i,h] + u[j,h]) * V[h]
  for (int o = lane; o < 64; o += 32) {
    const int ii = o >> 3, jj = o & 7;
    float acc = 0.f;
    for (int h = 0; h < 64; ++h) acc += tanhf(sa[w][ii * 64 + h] + su[w][jj * 64 + h]) * sV[h];
    sE[w][o] = acc;
  }
  __syncthreads();

  // softmax over j
  if (lane < 8) {
    float m = -1e30f;
    for (int j = 0; j < 8; ++j) m = fmaxf(m, sE[w][lane * 8 + j]);
    float ex[8], s = 0.f;
    for (int j = 0; j < 8; ++j) { ex[j] = __expf(sE[w][lane * 8 + j] - m); s += ex[j]; }
    const float inv = 1.f / s;
    for (int j = 0; j < 8; ++j) sAl[w][lane * 8 + j] = ex[j] * inv;
  }
  __syncthreads();

  // ctx[i,f] = sum_j alpha[i,j] X[j,f]  -  alpha[i,i] X[i,f]
  for (int i = lane; i < 512; i += 32) {
    const int t = i >> 6, f = i & 63;
    float c = 0.f;
    for (int j = 0; j < 8; ++j) c += sAl[w][t * 8 + j] * sX[w][j * 64 + f];
    c -= sAl[w][t * 8 + t] * sX[w][t * 64 + f];
    att_h[((size_t)b * 8 + t) * 128 + f]      = (_Float16)sX[w][t * 64 + f];
    att_h[((size_t)b * 8 + t) * 128 + 64 + f] = (_Float16)c;
  }
}

// ---------------------------------------------------------------------------
// 2) fc1_W (65536x256 f32, [K][N]) -> Wt (f16, [N][K]) via LDS tile transpose
// ---------------------------------------------------------------------------
__global__ void k_w1t(const float* __restrict__ W1, _Float16* __restrict__ Wt) {
  __shared__ _Float16 tile[64][65];
  const int kb = (blockIdx.x >> 2) << 6;  // 1024 k-tiles
  const int nb = (blockIdx.x & 3) << 6;   // 4 n-tiles
  const int tid = threadIdx.x;
  for (int i = tid; i < 64 * 64; i += 256) {
    const int kl = i >> 6, nl = i & 63;   // coalesced read over n
    tile[kl][nl] = (_Float16)W1[(size_t)(kb + kl) * HID1 + (nb + nl)];
  }
  __syncthreads();
  for (int i = tid; i < 64 * 64; i += 256) {
    const int nl = i >> 6, kl = i & 63;   // coalesced write over k
    Wt[(size_t)(nb + nl) * FLAT + (kb + kl)] = tile[kl][nl];
  }
}

// ---------------------------------------------------------------------------
// 3) graph path (tiny): degrees, two normalized-aggregate graph convs,
//    edge MLP, node_feats.
// ---------------------------------------------------------------------------
__global__ void k_degree(const int* __restrict__ src, const int* __restrict__ dst,
                         float* __restrict__ deg_out, float* __restrict__ deg_in) {
  int e = blockIdx.x * blockDim.x + threadIdx.x;
  if (e < BB) {
    atomicAdd(&deg_out[src[e]], 1.f);
    atomicAdd(&deg_in[dst[e]], 1.f);
  }
}

__global__ void k_gc_agg(const float* __restrict__ h, const int* __restrict__ src,
                         const int* __restrict__ dst, const float* __restrict__ deg_out,
                         float* __restrict__ agg) {
  const int e = blockIdx.x, f = threadIdx.x;       // 128 threads
  const int s = src[e], d = dst[e];
  const float sc = rsqrtf(fmaxf(deg_out[s], 1.f));
  atomicAdd(&agg[(size_t)d * DE + f], h[(size_t)s * DE + f] * sc);
}

__global__ void k_gc_fin1(const float* __restrict__ agg, const float* __restrict__ deg_in,
                          const float* __restrict__ Wg, const float* __restrict__ bg,
                          float* __restrict__ h1) {
  __shared__ float row[DE];
  const int n = blockIdx.x, j = threadIdx.x;       // 128 threads
  row[j] = agg[(size_t)n * DE + j] * rsqrtf(fmaxf(deg_in[n], 1.f));
  __syncthreads();
  float acc = bg[j];
  for (int f = 0; f < DE; ++f) acc += row[f] * Wg[f * 128 + j];
  h1[(size_t)n * 128 + j] = fmaxf(acc, 0.f);
}

__global__ void k_gc_fin2(const float* __restrict__ agg, const float* __restrict__ deg_in,
                          const float* __restrict__ Wg, const float* __restrict__ bg,
                          float* __restrict__ h2) {
  __shared__ float row[DE];
  const int n = blockIdx.x, j = threadIdx.x;       // 128 threads
  row[j] = agg[(size_t)n * DE + j] * rsqrtf(fmaxf(deg_in[n], 1.f));
  __syncthreads();
  if (j < 8) {
    float acc = bg[j];
    for (int f = 0; f < DE; ++f) acc += row[f] * Wg[f * 8 + j];
    h2[n * 8 + j] = acc;                           // no relu on h2
  }
}

__global__ void k_edge(const float* __restrict__ ef, const float* __restrict__ W1,
                       const float* __restrict__ b1, const float* __restrict__ W2,
                       const float* __restrict__ b2, const float* __restrict__ h2,
                       const int* __restrict__ src, const int* __restrict__ dst,
                       float* __restrict__ node_feats) {
  __shared__ float sef[DE], se1[DE];
  const int b = blockIdx.x, j = threadIdx.x;       // 128 threads
  sef[j] = ef[(size_t)b * DE + j];
  __syncthreads();
  float acc = b1[j];
  for (int f = 0; f < DE; ++f) acc += sef[f] * W1[f * 128 + j];
  se1[j] = fmaxf(acc, 0.f);
  __syncthreads();
  if (j < 8) {
    float a2 = b2[j];
    for (int f = 0; f < DE; ++f) a2 += se1[f] * W2[f * 8 + j];
    a2 = fmaxf(a2, 0.f);
    const int s = src[b], d = dst[b];
    const float hs = h2[s * 8 + j], hd = h2[d * 8 + j];
    node_feats[b * 24 + j]      = hs;
    node_feats[b * 24 + 8 + j]  = hd;
    node_feats[b * 24 + 16 + j] = hs + hd + a2;
  }
}

// ---------------------------------------------------------------------------
// 4) THE kernel: fused conv(2x2,SAME)+ReLU producer feeding WMMA fc1 GEMM.
//    Grid: 256 blocks (M-tile 16), 256 threads = 8 waves; wave w owns
//    N-columns [w*32, w*32+32) as two 16x16 accumulator tiles.
//    K = 65536 walked as 512 chunks of 128 (one (channel,t) pair each);
//    each chunk's 16x128 conv slab is computed once into double-buffered LDS.
// ---------------------------------------------------------------------------
#define MT 16
__global__ void __launch_bounds__(256) k_conv_fc1(
    const _Float16* __restrict__ att_h,   // [B][8][128]
    const float* __restrict__ conv_w,     // [64][1][2][2]
    const float* __restrict__ conv_b,     // [64]
    const _Float16* __restrict__ Wt,      // [256][65536] f16
    const float* __restrict__ fc1_b,      // [256]
    float* __restrict__ out1)             // [B][256]
{
  __shared__ __attribute__((aligned(16))) _Float16 satt[MT][9][132]; // zero-padded t=8,f=128
  __shared__ __attribute__((aligned(16))) _Float16 sconv[2][MT][136]; // 136-stride: DS-conflict-free
  __shared__ float scw[256];
  __shared__ float scb[64];

  const int tid = threadIdx.x;
  const int bbase = blockIdx.x * MT;

  if (tid < 256) scw[tid] = conv_w[tid];
  if (tid < 64)  scb[tid] = conv_b[tid];

  // stage att tile (f16) with zero padding for SAME conv (pad hi only)
  for (int i = tid; i < MT * 9 * 132; i += 256) {
    const int r = i / (9 * 132);
    const int rem = i - r * (9 * 132);
    const int t = rem / 132, f = rem - t * 132;
    _Float16 v = (_Float16)0.f;
    if (t < 8 && f < 128)
      v = att_h[(((size_t)(bbase + r)) * 8 + t) * 128 + f];
    satt[r][t][f] = v;
  }
  __syncthreads();

  const int wave   = tid >> 5;
  const int lane   = tid & 31;
  const int laneHi = lane >> 4;   // 0/1: half-wave selector in WMMA layouts
  const int laneLo = lane & 15;

  v8f acc0 = {}; v8f acc1 = {};
  const int col0 = wave * 32 + laneLo;        // global N col for b-frag 0
  const int col1 = wave * 32 + 16 + laneLo;   // global N col for b-frag 1
  const _Float16* wrow0 = Wt + (size_t)col0 * FLAT;
  const _Float16* wrow1 = Wt + (size_t)col1 * FLAT;

  for (int ch = 0; ch < 512; ++ch) {
    const int c = ch >> 3, t = ch & 7;
    const int buf = ch & 1;

    // produce conv chunk: 16 rows x 128 f values, ReLU'd, f16
    {
      const float w00 = scw[c * 4 + 0], w01 = scw[c * 4 + 1];
      const float w10 = scw[c * 4 + 2], w11 = scw[c * 4 + 3];
      const float bb = scb[c];
      for (int i = tid; i < MT * 128; i += 256) {
        const int r = i >> 7, f = i & 127;
        float v = bb
          + w00 * (float)satt[r][t][f]     + w01 * (float)satt[r][t][f + 1]
          + w10 * (float)satt[r][t + 1][f] + w11 * (float)satt[r][t + 1][f + 1];
        sconv[buf][r][f] = (_Float16)fmaxf(v, 0.f);
      }
    }
    __syncthreads();  // single barrier per chunk (double-buffered producer)

    // prefetch next chunk's W rows into L2/L0
    if (ch + 1 < 512) {
      __builtin_prefetch(wrow0 + (size_t)(ch + 1) * 128 + laneHi * 64, 0, 0);
      __builtin_prefetch(wrow1 + (size_t)(ch + 1) * 128 + laneHi * 64, 0, 0);
    }

    const size_t kbase = (size_t)ch * 128;
    #pragma unroll
    for (int kk = 0; kk < 4; ++kk) {
      // A 16x32 f16 fragment (row = laneLo):
      //   lanes 0-15  hold K = kk*32 + {0..15}
      //   lanes 16-31 hold K = kk*32 + {16..31}
      const int f0 = kk * 32 + laneHi * 16;
      v8h a_lo = *(const v8h*)&sconv[buf][laneLo][f0];
      v8h a_hi = *(const v8h*)&sconv[buf][laneLo][f0 + 8];
      v16h afrag = __builtin_shufflevector(a_lo, a_hi,
                     0, 1, 2, 3, 4, 5, 6, 7, 8, 9, 10, 11, 12, 13, 14, 15);
      // B 32x16 f16 fragment (col = laneLo): lanes 0-15 K 0..15, lanes 16-31 K 16..31
      v16h b0 = *(const v16h*)(wrow0 + kbase + kk * 32 + laneHi * 16);
      v16h b1 = *(const v16h*)(wrow1 + kbase + kk * 32 + laneHi * 16);
      acc0 = __builtin_amdgcn_wmma_f32_16x16x32_f16(false, afrag, false, b0,
                                                    (short)0, acc0, false, false);
      acc1 = __builtin_amdgcn_wmma_f32_16x16x32_f16(false, afrag, false, b1,
                                                    (short)0, acc1, false, false);
    }
  }

  // epilogue: D layout -> VGPR r holds M = r + 8*laneHi, N = laneLo; + bias, ReLU
  const float bias0 = fc1_b[col0];
  const float bias1 = fc1_b[col1];
  #pragma unroll
  for (int r = 0; r < 8; ++r) {
    const int m = bbase + laneHi * 8 + r;
    out1[(size_t)m * HID1 + col0] = fmaxf(acc0[r] + bias0, 0.f);
    out1[(size_t)m * HID1 + col1] = fmaxf(acc1[r] + bias1, 0.f);
  }
}

// ---------------------------------------------------------------------------
// 5) head: fc2 + ReLU, concat node_feats, fc3. One wave per batch row.
// ---------------------------------------------------------------------------
__global__ void k_head(const float* __restrict__ out1, const float* __restrict__ fc2W,
                       const float* __restrict__ fc2b, const float* __restrict__ nf,
                       const float* __restrict__ fc3W, const float* __restrict__ fc3b,
                       float* __restrict__ out) {
  __shared__ float buf[8][48];
  const int wave = threadIdx.x >> 5, lane = threadIdx.x & 31;
  const int b = blockIdx.x * 8 + wave;
  if (lane < 24) {
    float acc = fc2b[lane];
    const float* row = out1 + (size_t)b * HID1;
    for (int k = 0; k < HID1; ++k) acc += row[k] * fc2W[k * 24 + lane];
    buf[wave][lane] = fmaxf(acc, 0.f);
    buf[wave][24 + lane] = nf[b * 24 + lane];
  }
  __syncthreads();
  if (lane < 2) {
    float acc = fc3b[lane];
    for (int k = 0; k < 48; ++k) acc += buf[wave][k] * fc3W[k * 2 + lane];
    out[b * 2 + lane] = acc;
  }
}

// ---------------------------------------------------------------------------
extern "C" void kernel_launch(void* const* d_in, const int* in_sizes, int n_in,
                              void* d_out, int out_size, void* d_ws, size_t ws_size,
                              hipStream_t stream) {
  const float* X     = (const float*)d_in[0];
  const int*   src   = (const int*)d_in[1];
  const int*   dst   = (const int*)d_in[2];
  const float* nodef = (const float*)d_in[3];
  const float* edgef = (const float*)d_in[4];
  const float* attW  = (const float*)d_in[5];
  const float* attU  = (const float*)d_in[6];
  const float* attV  = (const float*)d_in[7];
  const float* convw = (const float*)d_in[8];
  const float* convb = (const float*)d_in[9];
  const float* gc1W  = (const float*)d_in[10];
  const float* gc1b  = (const float*)d_in[11];
  const float* gc2W  = (const float*)d_in[12];
  const float* gc2b  = (const float*)d_in[13];
  const float* lin1W = (const float*)d_in[14];
  const float* lin1b = (const float*)d_in[15];
  const float* lin2W = (const float*)d_in[16];
  const float* lin2b = (const float*)d_in[17];
  const float* fc1W  = (const float*)d_in[18];
  const float* fc1b  = (const float*)d_in[19];
  const float* fc2W  = (const float*)d_in[20];
  const float* fc2b  = (const float*)d_in[21];
  const float* fc3W  = (const float*)d_in[22];
  const float* fc3b  = (const float*)d_in[23];
  float* out = (float*)d_out;

  char* ws = (char*)d_ws;
  size_t off = 0;
  auto take = [&](size_t bytes) -> void* {
    off = (off + 255) & ~((size_t)255);
    void* p = (void*)(ws + off);
    off += bytes;
    return p;
  };

  _Float16* att_h = (_Float16*)take((size_t)BB * 8 * 128 * 2);   // 8 MB
  _Float16* Wt    = (_Float16*)take((size_t)HID1 * FLAT * 2);    // 32 MB
  float* out1     = (float*)take((size_t)BB * HID1 * 4);         // 4 MB
  const int nzero = 2048 + 2048 + NN * DE + NN * DE;             // degs + agg1 + agg2
  float* zbase    = (float*)take((size_t)nzero * 4);
  float* deg_out  = zbase;
  float* deg_in   = zbase + 2048;
  float* agg1     = zbase + 4096;
  float* agg2     = agg1 + NN * DE;
  float* h1       = (float*)take((size_t)NN * 128 * 4);
  float* h2       = (float*)take((size_t)NN * 8 * 4);
  float* nfeats   = (float*)take((size_t)BB * 24 * 4);

  // side buffers used with atomics must be zeroed every call
  k_zero<<<(nzero + 255) / 256, 256, 0, stream>>>(zbase, nzero);

  // attention -> att_out(f16)
  k_attention<<<BB / AW, 128, 0, stream>>>(X, attW, attU, attV, att_h);

  // fc1_W -> f16 [N][K]
  k_w1t<<<(FLAT / 64) * (HID1 / 64), 256, 0, stream>>>(fc1W, Wt);

  // graph path
  k_degree<<<(BB + 255) / 256, 256, 0, stream>>>(src, dst, deg_out, deg_in);
  k_gc_agg<<<BB, 128, 0, stream>>>(nodef, src, dst, deg_out, agg1);
  k_gc_fin1<<<NN, 128, 0, stream>>>(agg1, deg_in, gc1W, gc1b, h1);
  k_gc_agg<<<BB, 128, 0, stream>>>(h1, src, dst, deg_out, agg2);
  k_gc_fin2<<<NN, 128, 0, stream>>>(agg2, deg_in, gc2W, gc2b, h2);
  k_edge<<<BB, 128, 0, stream>>>(edgef, lin1W, lin1b, lin2W, lin2b, h2, src, dst, nfeats);

  // fused conv + fc1 (WMMA)
  k_conv_fc1<<<BB / MT, 256, 0, stream>>>(att_h, convw, convb, Wt, fc1b, out1);

  // head
  k_head<<<BB / 8, 256, 0, stream>>>(out1, fc2W, fc2b, nfeats, fc3W, fc3b, out);
}
